// DownLayer_48369921687850
// MI455X (gfx1250) — matrix operand order repaired and checked by reading, not compiled, not run on hardware
//
#include <hip/hip_runtime.h>
#include <hip/hip_bf16.h>

typedef __attribute__((ext_vector_type(16))) _Float16 v16h;
typedef __attribute__((ext_vector_type(8)))  float    v8f;

// Problem constants
#define BB     16
#define CIN    96
#define COUT   192
#define NIN    16384
#define NOUT   4096
#define KNB    16      // neighbors
#define MH     5       // harmonics
#define CB     (CIN*BB)        // 1536 columns of xT per j
#define NCH_CONV 15            // 480 / 32
#define NCH_POOL 3             // 96 / 32
#define NCH      18
#define CHUNK_E  512           // 32 lanes * 16 halfs per 32x16 f16 B-chunk
#define PANEL_N  (NCH*CHUNK_E) // 9216 f16 per n

// ---------------------------------------------------------------------------
// 1) Transpose x (B,CIN,NIN) f32  ->  xT[j][c*16+b] f16  (LDS-tiled, coalesced)
// ---------------------------------------------------------------------------
__global__ void k_transpose(const float* __restrict__ x, _Float16* __restrict__ xT) {
  __shared__ float tile[16][17];
  int j0 = blockIdx.x * 16;
  int p0 = blockIdx.y * 16;           // p = c*16 + b
  int tx = threadIdx.x, ty = threadIdx.y;
  int p = p0 + ty;
  int c = p >> 4, b = p & 15;
  tile[ty][tx] = x[(size_t)(b * CIN + c) * NIN + (j0 + tx)];
  __syncthreads();
  // write: thread (tx,ty) -> xT[(j0+ty)*CB + p0+tx]
  xT[(size_t)(j0 + ty) * CB + (p0 + tx)] = (_Float16)tile[tx][ty];
}

// ---------------------------------------------------------------------------
// 2) Repack weights into WMMA A-matrix register order (16x32 f16 tiles).
//    A layout (ISA 7.12.2): lane = 16*((k>>3)&1) + M ; elem = (k&7) + (k>>4)*8
// ---------------------------------------------------------------------------
__global__ void k_prepw(const float* __restrict__ conv_w, const float* __restrict__ pool_w,
                        _Float16* __restrict__ Aconv, _Float16* __restrict__ Aident) {
  const int total1 = COUT * 480;
  const int total2 = COUT * CIN;
  for (int e = blockIdx.x * blockDim.x + threadIdx.x; e < total1 + total2;
       e += gridDim.x * blockDim.x) {
    if (e < total1) {
      int o = e / 480, cm = e % 480;
      int ot = o >> 4, Mr = o & 15, chunk = cm >> 5, k = cm & 31;
      int lane = ((k >> 3) & 1) * 16 + Mr;
      int f    = (k & 7) + ((k >> 4) << 3);
      Aconv[((ot * NCH_CONV + chunk) * 32 + lane) * 16 + f] = (_Float16)conv_w[e];
    } else {
      int e2 = e - total1;
      int o = e2 / CIN, c = e2 % CIN;
      int ot = o >> 4, Mr = o & 15, chunk = c >> 5, k = c & 31;
      int lane = ((k >> 3) & 1) * 16 + Mr;
      int f    = (k & 7) + ((k >> 4) << 3);
      Aident[((ot * NCH_POOL + chunk) * 32 + lane) * 16 + f] = (_Float16)pool_w[e2];
    }
  }
}

// ---------------------------------------------------------------------------
// 3) Per-n feature panel: gather 16 neighbor rows of xT, weight by basis,
//    emit 576x16 B-panel (rows 0..479 conv feat, 480..575 pooled/16) in WMMA
//    B-matrix register order, staged in LDS then flushed coalesced.
//    B layout: lane = 16*((klocal>>3)&1) + col ; elem = (klocal&7) + (klocal>>4)*8
// ---------------------------------------------------------------------------
__global__ void k_panels(const _Float16* __restrict__ xT, const int* __restrict__ idx,
                         const float* __restrict__ basis, _Float16* __restrict__ panel) {
  int n = blockIdx.x;
  __shared__ int   jj[KNB];
  __shared__ float bas[KNB][MH];
  __shared__ alignas(16) _Float16 sp[PANEL_N];
  int t = threadIdx.x;
  if (t < KNB) jj[t] = idx[n * KNB + t];
  if (t < KNB * MH) bas[t / MH][t % MH] = basis[n * (KNB * MH) + t];
  __syncthreads();

  int b = t & 15, cg = t >> 4;          // 256 threads: 16 b-lanes x 16 c-groups
  for (int i = 0; i < 6; ++i) {         // 96 = 16 groups * 6
    int c = cg + 16 * i;
    float a0 = 0, a1 = 0, a2 = 0, a3 = 0, a4 = 0, pool = 0;
    const _Float16* base = xT + c * 16 + b;
#pragma unroll
    for (int k = 0; k < KNB; ++k) {
      float v = (float)base[(size_t)jj[k] * CB];
      pool += v;
      a0 += v * bas[k][0]; a1 += v * bas[k][1]; a2 += v * bas[k][2];
      a3 += v * bas[k][3]; a4 += v * bas[k][4];
    }
    float acc[MH] = {a0, a1, a2, a3, a4};
#pragma unroll
    for (int m = 0; m < MH; ++m) {
      int r = c * MH + m;                   // conv row 0..479
      int chunk = r >> 5, kl = r & 31;
      sp[(chunk * 32 + ((kl >> 3) & 1) * 16 + b) * 16 + (kl & 7) + ((kl >> 4) << 3)] = (_Float16)acc[m];
    }
    int chunk2 = NCH_CONV + (c >> 5), kl2 = c & 31;   // pooled row 480+c
    sp[(chunk2 * 32 + ((kl2 >> 3) & 1) * 16 + b) * 16 + (kl2 & 7) + ((kl2 >> 4) << 3)] = (_Float16)(pool * 0.0625f);
  }
  __syncthreads();
  const uint4* src = (const uint4*)sp;                 // 18432 B = 1152 uint4
  uint4* dst = (uint4*)(panel + (size_t)n * PANEL_N);
  for (int e = t; e < 1152; e += 256) dst[e] = src[e];
}

// ---------------------------------------------------------------------------
// 4) Pass 1: conv GEMM tiles -> per-block per-channel sum / sum^2 partials.
//    384 threads = 12 waves; wave w owns o-tile w. B panel staged in LDS once
//    (no register-array hoisting -> no scratch spills in the WMMA loop).
// ---------------------------------------------------------------------------
__global__ void k_conv_stats(const _Float16* __restrict__ panel,
                             const _Float16* __restrict__ Aconv,
                             float* __restrict__ partial) {
  int n = blockIdx.x;
  int t = threadIdx.x, lane = t & 31, ot = t >> 5;   // 12 waves
  __shared__ alignas(16) _Float16 sp[PANEL_N];
  {
    const uint4* src = (const uint4*)(panel + (size_t)n * PANEL_N);
    uint4* dst = (uint4*)sp;
    for (int e = t; e < PANEL_N / 8; e += 384) dst[e] = src[e];
  }
  __syncthreads();
  const v16h* Bp = (const v16h*)sp;
  const v16h* Ap = (const v16h*)(Aconv + (size_t)ot * NCH_CONV * CHUNK_E);
  v8f acc = {};
#pragma unroll
  for (int ch = 0; ch < NCH_CONV; ++ch) {
    v16h a  = Ap[ch * 32 + lane];
    v16h bm = Bp[ch * 32 + lane];
    acc = __builtin_amdgcn_wmma_f32_16x16x32_f16(false, a, false, bm,
                                                 (short)0, acc, false, false);
  }
  float* P = partial + (size_t)n * (2 * COUT);
#pragma unroll
  for (int i = 0; i < 8; ++i) {
    float v = acc[i], v2 = v * v;
    for (int off = 8; off > 0; off >>= 1) {   // reduce across the 16 columns
      v  += __shfl_down(v,  off, 16);
      v2 += __shfl_down(v2, off, 16);
    }
    if ((lane & 15) == 0) {
      int ch = ot * 16 + ((lane >> 4) << 3) + i;   // C layout: M = i (+8 for hi half)
      P[ch]        = v;
      P[COUT + ch] = v2;
    }
  }
}

// ---------------------------------------------------------------------------
// 5) Deterministic tree reduction of the 4096 partials per stat-channel.
// ---------------------------------------------------------------------------
__global__ void k_reduce(const float* __restrict__ partial, float* __restrict__ stats) {
  int s = blockIdx.x;                       // 0..383  (sum, then sumsq)
  int t = threadIdx.x;
  __shared__ float red[256];
  float acc = 0.f;
  for (int k = t; k < NOUT; k += 256) acc += partial[(size_t)k * (2 * COUT) + s];
  red[t] = acc;
  __syncthreads();
  for (int off = 128; off > 0; off >>= 1) {
    if (t < off) red[t] += red[t + off];
    __syncthreads();
  }
  if (t == 0) stats[s] = red[0];
}

__global__ void k_finalize(const float* __restrict__ stats, const float* __restrict__ gamma,
                           const float* __restrict__ beta, const float* __restrict__ pool_b,
                           float* __restrict__ scale_shift) {
  int o = threadIdx.x;
  if (o < COUT) {
    const float inv = 1.0f / (float)(BB * NOUT);
    float mean = stats[o] * inv;
    float var  = stats[COUT + o] * inv - mean * mean;
    float sc   = gamma[o] * rsqrtf(var + 1e-5f);
    scale_shift[o]        = sc;
    scale_shift[COUT + o] = beta[o] - mean * sc + pool_b[o];  // conv_b cancels in BN
  }
}

// ---------------------------------------------------------------------------
// 6) Pass 2: recompute conv tile + identity tile (one o-tile per wave, B panel
//    in LDS), apply BN affine, write out.
// ---------------------------------------------------------------------------
__global__ void k_final(const _Float16* __restrict__ panel, const _Float16* __restrict__ Aconv,
                        const _Float16* __restrict__ Aident,
                        const float* __restrict__ scale_shift, float* __restrict__ out) {
  int n = blockIdx.x;
  int t = threadIdx.x, lane = t & 31, ot = t >> 5;   // 12 waves
  __shared__ alignas(16) _Float16 sp[PANEL_N];
  __shared__ float ssc[COUT], ssh[COUT];
  {
    const uint4* src = (const uint4*)(panel + (size_t)n * PANEL_N);
    uint4* dst = (uint4*)sp;
    for (int e = t; e < PANEL_N / 8; e += 384) dst[e] = src[e];
    if (t < COUT) { ssc[t] = scale_shift[t]; ssh[t] = scale_shift[COUT + t]; }
  }
  __syncthreads();
  const v16h* Bp = (const v16h*)sp;

  const v16h* Ap = (const v16h*)(Aconv + (size_t)ot * NCH_CONV * CHUNK_E);
  v8f acc = {};
#pragma unroll
  for (int ch = 0; ch < NCH_CONV; ++ch)
    acc = __builtin_amdgcn_wmma_f32_16x16x32_f16(false, Ap[ch * 32 + lane], false,
                                                 Bp[ch * 32 + lane], (short)0, acc,
                                                 false, false);
  const v16h* Ai = (const v16h*)(Aident + (size_t)ot * NCH_POOL * CHUNK_E);
  v8f acc2 = {};
#pragma unroll
  for (int ch = 0; ch < NCH_POOL; ++ch)
    acc2 = __builtin_amdgcn_wmma_f32_16x16x32_f16(false, Ai[ch * 32 + lane], false,
                                                  Bp[(NCH_CONV + ch) * 32 + lane], (short)0,
                                                  acc2, false, false);
  int b = lane & 15;
  int obase = ot * 16 + ((lane >> 4) << 3);
#pragma unroll
  for (int i = 0; i < 8; ++i) {
    int o = obase + i;
    out[(size_t)(b * COUT + o) * NOUT + n] = acc[i] * ssc[o] + ssh[o] + acc2[i];
  }
}

// ---------------------------------------------------------------------------
// Workspace layout (bytes)
//   xT      : 0            .. 50331648   (16384*1536 f16)
//   panel   : 50331648     .. 125829120  (4096*9216 f16)
//   Aconv   : 125829120    .. 126013440  (12*15*512 f16)
//   Aident  : 126013440    .. 126050304  (12*3*512 f16)
//   partial : 126050304    .. 132341760  (4096*384 f32)
//   stats   : 132341760    .. +3072      (768 f32: sum, sumsq, scale, shift)
// ---------------------------------------------------------------------------
extern "C" void kernel_launch(void* const* d_in, const int* in_sizes, int n_in,
                              void* d_out, int out_size, void* d_ws, size_t ws_size,
                              hipStream_t stream) {
  (void)in_sizes; (void)n_in; (void)out_size; (void)ws_size;
  const float* x      = (const float*)d_in[0];
  const int*   idx    = (const int*)d_in[1];
  const float* basis  = (const float*)d_in[2];
  const float* conv_w = (const float*)d_in[3];
  /* d_in[4] = conv_b : cancels through BN mean-subtraction */
  const float* pool_w = (const float*)d_in[5];
  const float* pool_b = (const float*)d_in[6];
  const float* gamma  = (const float*)d_in[7];
  const float* beta   = (const float*)d_in[8];

  char* ws = (char*)d_ws;
  _Float16* xT      = (_Float16*)(ws);
  _Float16* panel   = (_Float16*)(ws + 50331648);
  _Float16* Aconv   = (_Float16*)(ws + 125829120);
  _Float16* Aident  = (_Float16*)(ws + 126013440);
  float*    partial = (float*)   (ws + 126050304);
  float*    stats   = (float*)   (ws + 132341760);
  float*    out     = (float*)d_out;

  k_transpose <<<dim3(NIN / 16, CB / 16), dim3(16, 16), 0, stream>>>(x, xT);
  k_prepw     <<<dim3(64),   dim3(256), 0, stream>>>(conv_w, pool_w, Aconv, Aident);
  k_panels    <<<dim3(NOUT), dim3(256), 0, stream>>>(xT, idx, basis, panel);
  k_conv_stats<<<dim3(NOUT), dim3(384), 0, stream>>>(panel, Aconv, partial);
  k_reduce    <<<dim3(2 * COUT), dim3(256), 0, stream>>>(partial, stats);
  k_finalize  <<<dim3(1),    dim3(192), 0, stream>>>(stats, gamma, beta, pool_b, stats + 384);
  k_final     <<<dim3(NOUT), dim3(384), 0, stream>>>(panel, Aconv, Aident, stats + 384, out);
}
